// SupConLoss_86242943304233
// MI455X (gfx1250) — compile-verified, matrix operand order checked by reading, not run on hardware
//
#include <hip/hip_runtime.h>
#include <hip/hip_bf16.h>

typedef __attribute__((ext_vector_type(16))) _Float16 v16h;
typedef __attribute__((ext_vector_type(8)))  _Float16 v8h;
typedef __attribute__((ext_vector_type(8)))  float    v8f;
typedef __attribute__((ext_vector_type(4)))  int      v4i;

#define BATCH 8192
#define DIM   256
#define INV_T 14.285714285714286f   /* 1/0.07 */
#define LDSROW 264                   /* 256 + 8-half pad: 528B stride, bank-conflict free */

#define AS1 __attribute__((address_space(1)))
#define AS3 __attribute__((address_space(3)))

#if __has_builtin(__builtin_amdgcn_global_load_async_to_lds_b128) && \
    __has_builtin(__builtin_amdgcn_s_wait_asynccnt)
#define HAS_ASYNC 1
#else
#define HAS_ASYNC 0
#endif

// ---------------------------------------------------------------------------
// Kernel 1: row-wise L2 normalize fp32 -> f16 (matches F.normalize eps clamp)
// ---------------------------------------------------------------------------
__global__ __launch_bounds__(DIM) void k_normalize(const float* __restrict__ f,
                                                   _Float16* __restrict__ nf) {
  __shared__ float sred[DIM];
  const int r = blockIdx.x, t = threadIdx.x;
  float v = f[r * DIM + t];
  sred[t] = v * v;
  __syncthreads();
#pragma unroll
  for (int s = DIM / 2; s > 0; s >>= 1) {
    if (t < s) sred[t] += sred[t + s];
    __syncthreads();
  }
  const float scale = 1.0f / fmaxf(sqrtf(sred[0]), 1e-12f);
  nf[r * DIM + t] = (_Float16)(v * scale);
}

// Build a v16h WMMA fragment from two 16B chunks (ISA 7.12.2 16-bit layout:
// low lanes K 0-7 & 16-23, high lanes K 8-15 & 24-31; identical for A and B
// here since B = feat^T).
__device__ __forceinline__ v16h load_frag(const _Float16* p) {
  v8h lo = *(const v8h*)(p);
  v8h hi = *(const v8h*)(p + 16);
  return __builtin_shufflevector(lo, hi, 0, 1, 2, 3, 4, 5, 6, 7,
                                 8, 9, 10, 11, 12, 13, 14, 15);
}

// Stage one 64-col x 256-K f16 chunk (32KB, contiguous in global) into LDS
// with padded rows. 256 threads x 128B each. Async (ASYNCcnt) when available.
__device__ __forceinline__ void stage_chunk(const _Float16* __restrict__ gsrc,
                                            _Float16* lbase, int tid) {
  const int col  = tid >> 2;         // 0..63
  const int part = (tid & 3) * 64;   // halves within the row
  const _Float16* g = gsrc + col * DIM + part;
  _Float16*       l = lbase + col * LDSROW + part;
#if HAS_ASYNC
#pragma unroll
  for (int i = 0; i < 8; ++i)        // 8 x b128 per thread
    __builtin_amdgcn_global_load_async_to_lds_b128(
        (AS1 v4i*)(g + i * 8), (AS3 v4i*)(l + i * 8), 0, 0);
#else
#pragma unroll
  for (int i = 0; i < 8; ++i)
    *(v8h*)(l + i * 8) = *(const v8h*)(g + i * 8);
#endif
}

__device__ __forceinline__ void wait_stage() {
#if HAS_ASYNC
  __builtin_amdgcn_s_wait_asynccnt(0);
#endif
}

// ---------------------------------------------------------------------------
// Kernel 2: fused feat@feat^T (WMMA f16->f32) + masked logsumexp reductions.
// Block = 256 threads = 8 wave32s, owns a 32-row strip; B tiles double-
// buffered through LDS via async global->LDS copies.
// Row-max subtraction folds exactly to l' = (sim - 1)/T (diag == 1).
// ---------------------------------------------------------------------------
__global__ __launch_bounds__(256) void k_supcon(const _Float16* __restrict__ nf,
                                                const int* __restrict__ labels,
                                                float* __restrict__ partial) {
  __shared__ _Float16 sB[2][64 * LDSROW];   // 2 x 33KB B-chunk buffers
  __shared__ float sS[32][4];
  __shared__ float sZ[32][4];
  __shared__ float sN[32][4];
  __shared__ float sT[32];

  const int tid   = threadIdx.x;
  const int wave  = tid >> 5;
  const int lane  = tid & 31;
  const int mtile = wave & 1;        // 2 row tiles of 16
  const int ntile = wave >> 1;       // 4 col tiles of 16 per 64-col chunk
  const int Ibase = blockIdx.x * 32;
  const int khalf = (lane >> 4) * 8; // K split across lane halves

  // --- A fragments: this wave's 16 rows x K=256, resident in VGPRs ---
  const int rowA = Ibase + mtile * 16 + (lane & 15);
  const _Float16* arow = nf + rowA * DIM;
  v16h a[8];
#pragma unroll
  for (int kb = 0; kb < 8; ++kb)
    a[kb] = load_frag(arow + kb * 32 + khalf);

  // labels of the 8 M-rows this lane accumulates (C layout: VGPR r ->
  // M = r + 8*(lane/16), N = lane&15)
  int labR[8];
  const int rbase = Ibase + mtile * 16 + (lane >> 4) * 8;
#pragma unroll
  for (int r = 0; r < 8; ++r) labR[r] = labels[rbase + r];

  float accS[8], accZ[8], accN[8];
#pragma unroll
  for (int r = 0; r < 8; ++r) { accS[r] = 0.f; accZ[r] = 0.f; accN[r] = 0.f; }

  const int colLocal = ntile * 16 + (lane & 15);

  // --- software-pipelined sweep over all 8192 columns, 64 per chunk ---
  stage_chunk(nf, &sB[0][0], tid);                       // prologue: chunk 0
  for (int i = 0; i < BATCH / 64; ++i) {
    wait_stage();                                        // my async copies done
    __syncthreads();                                     // everyone's done
    if (i + 1 < BATCH / 64)                              // overlap next chunk
      stage_chunk(nf + (size_t)(i + 1) * 64 * DIM, &sB[(i + 1) & 1][0], tid);

    const int labC = labels[i * 64 + colLocal];
    const _Float16* brow = &sB[i & 1][0] + colLocal * LDSROW;

    v8f c = {};
#pragma unroll
    for (int kb = 0; kb < 8; ++kb) {
      v16h b = load_frag(brow + kb * 32 + khalf);        // ds_load_b128 x2
      c = __builtin_amdgcn_wmma_f32_16x16x32_f16(
              /*neg_a=*/false, a[kb], /*neg_b=*/false, b,
              /*c_mod=*/(short)0, c, /*reuse_a=*/false, /*reuse_b=*/false);
    }

    // epilogue: masked accumulate per row (branchless)
#pragma unroll
    for (int r = 0; r < 8; ++r) {
      const float lp = (c[r] - 1.0f) * INV_T;            // shifted logit, <= 0
      const float mk = (labR[r] == labC) ? 1.0f : 0.0f;
      accS[r] += mk * lp;
      accZ[r] += mk * __expf(lp);
      accN[r] += mk;
    }
    __syncthreads();       // all reads of sB[i&1] done before reuse at i+2
  }

  // --- reduce across the 16 lanes of each half (N dimension) ---
#pragma unroll
  for (int r = 0; r < 8; ++r) {
#pragma unroll
    for (int msk = 1; msk < 16; msk <<= 1) {
      accS[r] += __shfl_xor(accS[r], msk, 32);
      accZ[r] += __shfl_xor(accZ[r], msk, 32);
      accN[r] += __shfl_xor(accN[r], msk, 32);
    }
  }

  // each wave writes its own slot -> no atomics, deterministic
  if ((lane & 15) == 0) {
    const int base = mtile * 16 + (lane >> 4) * 8;
#pragma unroll
    for (int r = 0; r < 8; ++r) {
      sS[base + r][ntile] = accS[r];
      sZ[base + r][ntile] = accZ[r];
      sN[base + r][ntile] = accN[r];
    }
  }
  __syncthreads();

  if (tid < 32) {
    float S = 0.f, Z = 0.f, N = 0.f;
#pragma unroll
    for (int t = 0; t < 4; ++t) { S += sS[tid][t]; Z += sZ[tid][t]; N += sN[tid][t]; }
    sT[tid] = S - N * __logf(Z);   // n_i >= 1 (diagonal), Z >= 1
  }
  __syncthreads();

  if (tid == 0) {
    float s = 0.f;
#pragma unroll
    for (int i = 0; i < 32; ++i) s += sT[i];
    partial[blockIdx.x] = s;
  }
}

// ---------------------------------------------------------------------------
// Kernel 3: deterministic fixed-order reduction of 256 block partials.
// ---------------------------------------------------------------------------
__global__ __launch_bounds__(256) void k_finalize(const float* __restrict__ partial,
                                                  float* __restrict__ out) {
  __shared__ float sred[256];
  const int t = threadIdx.x;
  sred[t] = partial[t];
  __syncthreads();
#pragma unroll
  for (int s = 128; s > 0; s >>= 1) {
    if (t < s) sred[t] += sred[t + s];
    __syncthreads();
  }
  if (t == 0) out[0] = -sred[0] / (float)BATCH;
}

// ---------------------------------------------------------------------------
extern "C" void kernel_launch(void* const* d_in, const int* in_sizes, int n_in,
                              void* d_out, int out_size, void* d_ws, size_t ws_size,
                              hipStream_t stream) {
  const float* features = (const float*)d_in[0];
  const int*   labels   = (const int*)d_in[1];
  float*       out      = (float*)d_out;

  // ws layout: [0, 4MB) normalized f16 features; [4MB, +1KB) block partials
  _Float16* nf      = (_Float16*)d_ws;
  float*    partial = (float*)((char*)d_ws + (size_t)BATCH * DIM * sizeof(_Float16));

  k_normalize<<<BATCH, DIM, 0, stream>>>(features, nf);
  k_supcon<<<BATCH / 32, 256, 0, stream>>>(nf, labels, partial);
  k_finalize<<<1, 256, 0, stream>>>(partial, out);
}
